// VectorQuantizer_37349035606504
// MI455X (gfx1250) — matrix operand order repaired and checked by reading, not compiled, not run on hardware
//
#include <hip/hip_runtime.h>

#define NATOMS 500000
#define NFEAT  9
#define EMB    300
#define NTOK   512
#define KPAD   320                  // 10 k-tiles of 32
#define NTILES (NATOMS / 16)        // 31250 waves, one 16-atom tile each
#define WPB    2                    // waves per block
#define NBLK   (NTILES / WPB)       // 15625 (exact)
#define NSLOTS NTILES               // 31250 partial-sum slots
#define ASLOT  48                   // LDS bytes per (ktile,lane) A slot (bank-perfect stride)
#define AMATB  (10 * 32 * ASLOT)    // 15360 B per (wave, hi|lo) A region

typedef __attribute__((ext_vector_type(16))) __bf16       v16bf;
typedef __attribute__((ext_vector_type(8)))  float        v8f;
typedef __attribute__((ext_vector_type(4)))  float        f32x4;
typedef __attribute__((ext_vector_type(4)))  unsigned int u32x4;

union V16U { v16bf v; unsigned short u[16]; u32x4 q[2]; };

__device__ __forceinline__ unsigned short f32_bf16(float f) {
  unsigned u = __float_as_uint(f);
  u += 0x7FFFu + ((u >> 16) & 1u);          // round-to-nearest-even truncation
  return (unsigned short)(u >> 16);
}
__device__ __forceinline__ float bf16_f32(unsigned short h) {
  return __uint_as_float(((unsigned)h) << 16);
}

// ---------------------------------------------------------------------------
// Prep: Wt_hi/Wt_lo[token][k] bf16 split, token-major (K padded with zeros) and
// per-token squared norms. Weight is tiny (614 KB) — this kernel is noise.
// ---------------------------------------------------------------------------
__global__ void vq_prep(const float* __restrict__ w,
                        unsigned short* __restrict__ wt_hi,
                        unsigned short* __restrict__ wt_lo,
                        float* __restrict__ wsq) {
  const int token = blockIdx.x;   // 512 blocks
  const int k     = threadIdx.x;  // 320 threads
  float v = (k < EMB) ? w[(size_t)token * EMB + k] : 0.f;
  unsigned short h = f32_bf16(v);
  unsigned short l = f32_bf16(v - bf16_f32(h));
  wt_hi[(size_t)token * KPAD + k] = h;
  wt_lo[(size_t)token * KPAD + k] = l;
  __shared__ float s[KPAD];
  s[k] = v * v;
  __syncthreads();
  if (k == 0) {
    float acc = 0.f;
    for (int i = 0; i < EMB; ++i) acc += s[i];
    wsq[token] = acc;
  }
}

// ---------------------------------------------------------------------------
// Main: one wave per 16-atom tile. A tiles staged in LDS (bf16 hi/lo), B
// streamed from global. Coltiles processed in chunks of 8 so the 8 live v8f
// accumulators + A/B operands stay well under the VGPR budget (no spills).
// ---------------------------------------------------------------------------
__global__ void __launch_bounds__(32 * WPB)
vq_main(const int*   __restrict__ xp,
        const float* __restrict__ e,
        const float* __restrict__ w,
        const unsigned short* __restrict__ wt_hi,
        const unsigned short* __restrict__ wt_lo,
        const float* __restrict__ wsq,
        float* __restrict__ outq,
        float* __restrict__ partials) {
  __shared__ alignas(16) unsigned char aLds[WPB * 2 * AMATB];   // 60 KB

  const int lane = threadIdx.x & 31;
  const int wave = threadIdx.x >> 5;
  const int tile = blockIdx.x * WPB + wave;      // grid divides exactly
  const int half = lane >> 4;                    // 0: C rows 0-7, 1: rows 8-15
  const int l16  = lane & 15;

  unsigned char* ahiB = aLds + (size_t)wave * 2 * AMATB;
  unsigned char* aloB = ahiB + AMATB;

  // Token-range bounds for the 8 C-matrix rows this lane owns.
  int lo_r[8], hi_r[8];
  #pragma unroll
  for (int j = 0; j < 8; ++j) {
    int m = j + 8 * half;
    int a = xp[(size_t)(tile * 16 + m) * NFEAT];
    lo_r[j] = (a == 5) ? 0   : (a == 6) ? 378 : (a == 7) ? 434 : 489;
    hi_r[j] = (a == 5) ? 377 : (a == 6) ? 433 : (a == 7) ? 488 : 511;
  }

  // Phase 1: load e tile in WMMA A-operand layout, bf16 hi/lo split, stage in
  // LDS. Lane l holds row (l&15); halves[0..7] = K kb..kb+7, halves[8..15] =
  // K kb+16..kb+23, kb = 8*(l>=16) — per the ISA 16-bit A 16x32 table.
  float esq_part = 0.f;
  const float* erow = e + (size_t)(tile * 16 + l16) * EMB;
  const int kb = 8 * half;
  #pragma unroll
  for (int t = 0; t < 10; ++t) {
    float x[16];
    if (t < 9) {                         // tiles 0..8: K <= 287 < 300, unguarded
      const f32x4* p1 = (const f32x4*)(erow + t * 32 + kb);
      const f32x4* p2 = (const f32x4*)(erow + t * 32 + kb + 16);
      f32x4 a0 = p1[0], a1 = p1[1], b0 = p2[0], b1 = p2[1];
      x[0]=a0.x;  x[1]=a0.y;  x[2]=a0.z;  x[3]=a0.w;
      x[4]=a1.x;  x[5]=a1.y;  x[6]=a1.z;  x[7]=a1.w;
      x[8]=b0.x;  x[9]=b0.y;  x[10]=b0.z; x[11]=b0.w;
      x[12]=b1.x; x[13]=b1.y; x[14]=b1.z; x[15]=b1.w;
    } else {                             // last tile: K 288..319, guard at 300
      #pragma unroll
      for (int i = 0; i < 8; ++i) { int k = 288 + kb + i;      x[i]     = (k < EMB) ? erow[k] : 0.f; }
      #pragma unroll
      for (int i = 0; i < 8; ++i) { int k = 288 + kb + 16 + i; x[8 + i] = (k < EMB) ? erow[k] : 0.f; }
    }
    V16U hh, ll;
    #pragma unroll
    for (int i = 0; i < 16; ++i) {
      esq_part += x[i] * x[i];
      unsigned short h = f32_bf16(x[i]);
      hh.u[i] = h;
      ll.u[i] = f32_bf16(x[i] - bf16_f32(h));
    }
    u32x4* ph = (u32x4*)(ahiB + (t * 32 + lane) * ASLOT);
    ph[0] = hh.q[0]; ph[1] = hh.q[1];
    u32x4* pl = (u32x4*)(aloB + (t * 32 + lane) * ASLOT);
    pl[0] = ll.q[0]; pl[1] = ll.q[1];
  }
  // Full ||e||^2 of row (lane&15): the two 16-lane halves hold disjoint K chunks.
  float esq_row = esq_part + __shfl_xor(esq_part, 16, 32);

  float bestd[8];
  int   bestn[8];
  #pragma unroll
  for (int j = 0; j < 8; ++j) { bestd[j] = __builtin_inff(); bestn[j] = 0; }

  // Phase 2: 4 chunks x (10 ktiles x 8 coltiles x 3 WMMAs).
  for (int chunk = 0; chunk < 4; ++chunk) {
    v8f acc[8];
    #pragma unroll
    for (int c = 0; c < 8; ++c) acc[c] = (v8f){};

    for (int t = 0; t < 10; ++t) {
      V16U ah, al;
      const u32x4* pah = (const u32x4*)(ahiB + (t * 32 + lane) * ASLOT);
      ah.q[0] = pah[0]; ah.q[1] = pah[1];
      const u32x4* pal = (const u32x4*)(aloB + (t * 32 + lane) * ASLOT);
      al.q[0] = pal[0]; al.q[1] = pal[1];

      // B operand: lane l holds token column (l&15), K kb..kb+7 and kb+16..kb+23 (same order as A).
      const unsigned short* bh0 = wt_hi + (size_t)(chunk * 128 + l16) * KPAD + t * 32 + kb;
      const unsigned short* bl0 = wt_lo + (size_t)(chunk * 128 + l16) * KPAD + t * 32 + kb;
      #pragma unroll
      for (int c = 0; c < 8; ++c) {
        V16U bh, bl;
        const u32x4* ph = (const u32x4*)(bh0 + (size_t)c * 16 * KPAD);
        const u32x4* pl = (const u32x4*)(bl0 + (size_t)c * 16 * KPAD);
        bh.q[0] = ph[0]; bh.q[1] = ph[2];
        bl.q[0] = pl[0]; bl.q[1] = pl[2];
        // dot ~= hi*hi + lo*hi + hi*lo (fp32 accumulate; lo*lo ~2^-16, dropped)
        acc[c] = __builtin_amdgcn_wmma_f32_16x16x32_bf16(false, ah.v, false, bh.v, (short)0, acc[c], false, false);
        acc[c] = __builtin_amdgcn_wmma_f32_16x16x32_bf16(false, al.v, false, bh.v, (short)0, acc[c], false, false);
        acc[c] = __builtin_amdgcn_wmma_f32_16x16x32_bf16(false, ah.v, false, bl.v, (short)0, acc[c], false, false);
      }
    }

    #pragma unroll
    for (int c = 0; c < 8; ++c) {
      const int   n  = (chunk * 8 + c) * 16 + l16;   // token of this lane's C column
      const float wn = wsq[n];
      #pragma unroll
      for (int j = 0; j < 8; ++j) {
        float d = wn - 2.f * acc[c][j];    // ||e||^2 omitted: row-constant for argmin
        bool valid = (n >= lo_r[j]) && (n < hi_r[j]);
        if (valid && (d < bestd[j])) { bestd[j] = d; bestn[j] = n; }
      }
    }
  }

  // Argmin across the 16 lanes of each half; ties -> smaller token index
  // (matches jnp.argmin first-occurrence semantics).
  #pragma unroll
  for (int j = 0; j < 8; ++j) {
    #pragma unroll
    for (int off = 1; off < 16; off <<= 1) {
      float od = __shfl_xor(bestd[j], off, 32);
      int   on = __shfl_xor(bestn[j], off, 32);
      if (od < bestd[j] || (od == bestd[j] && on < bestn[j])) { bestd[j] = od; bestn[j] = on; }
    }
  }

  // Loss partial: sum over 16 rows of (||e||^2 + best_d) == sum ||w_idx - e||^2.
  float sd = 0.f;
  #pragma unroll
  for (int j = 0; j < 8; ++j) sd += bestd[j];
  sd += __shfl_xor(sd, 16, 32);            // add the other half's 8 rows
  float se = esq_row;
  #pragma unroll
  for (int off = 1; off < 16; off <<= 1) se += __shfl_xor(se, off, 32);
  if (lane == 0) partials[tile] = sd + se;

  // Emit quantized rows: copy fp32 weight[idx] (L2-resident) cooperatively.
  #pragma unroll
  for (int m = 0; m < 16; ++m) {
    int idx = __shfl(bestn[m & 7], (m >> 3) * 16, 32);
    const f32x4* src = (const f32x4*)(w + (size_t)idx * EMB);
    f32x4*       dst = (f32x4*)(outq + (size_t)(tile * 16 + m) * EMB);
    for (int i = lane; i < EMB / 4; i += 32) dst[i] = src[i];   // 75 float4s, 16B aligned
  }
}

// ---------------------------------------------------------------------------
// Deterministic single-block reduction of per-wave partials -> scalar loss.
// ---------------------------------------------------------------------------
__global__ void vq_reduce(const float* __restrict__ partials, int n,
                          float* __restrict__ out_loss) {
  __shared__ double s[256];
  double acc = 0.0;
  for (int i = threadIdx.x; i < n; i += 256) acc += (double)partials[i];
  s[threadIdx.x] = acc;
  __syncthreads();
  for (int off = 128; off > 0; off >>= 1) {
    if (threadIdx.x < off) s[threadIdx.x] += s[threadIdx.x + off];
    __syncthreads();
  }
  if (threadIdx.x == 0)
    *out_loss = (float)(1.25 * s[0] / ((double)NATOMS * (double)EMB));
}

extern "C" void kernel_launch(void* const* d_in, const int* in_sizes, int n_in,
                              void* d_out, int out_size, void* d_ws, size_t ws_size,
                              hipStream_t stream) {
  const int*   xp = (const int*)d_in[0];    // (500000, 9) int32
  const float* e  = (const float*)d_in[1];  // (500000, 300) f32
  const float* w  = (const float*)d_in[2];  // (512, 300) f32
  float* outq = (float*)d_out;              // 500000*300 quantized, then 1 loss
  float* loss = outq + (size_t)NATOMS * EMB;

  char* ws = (char*)d_ws;                   // ~782 KB total
  unsigned short* wt_hi   = (unsigned short*)(ws);
  unsigned short* wt_lo   = (unsigned short*)(ws + (size_t)KPAD * NTOK * 2);
  float*          wsq     = (float*)(ws + (size_t)KPAD * NTOK * 4);
  float*          partial = (float*)(ws + (size_t)KPAD * NTOK * 4 + (size_t)NTOK * 4);

  vq_prep<<<NTOK, KPAD, 0, stream>>>(w, wt_hi, wt_lo, wsq);
  vq_main<<<NBLK, 32 * WPB, 0, stream>>>(xp, e, w, wt_hi, wt_lo, wsq, outq, partial);
  vq_reduce<<<1, 256, 0, stream>>>(partial, NSLOTS, loss);
}
